// Attention_81681688035903
// MI455X (gfx1250) — matrix operand order repaired and checked by reading, not compiled, not run on hardware
//
#include <hip/hip_runtime.h>

#define D_MODEL 1024
#define BATCH   8
#define SEQ     1024
#define HEADS   8
#define DHEAD   128
#define USE_ASYNC 1

#if defined(__has_builtin)
# if __has_builtin(__builtin_amdgcn_tensor_load_to_lds)
#  define HAVE_TDM 1
# else
#  define HAVE_TDM 0
# endif
#else
# define HAVE_TDM 0
#endif

typedef __bf16 bf16_t;
typedef __attribute__((ext_vector_type(16))) __bf16 v16bf;
typedef __attribute__((ext_vector_type(8)))  __bf16 v8bf;
typedef __attribute__((ext_vector_type(8)))  float  v8f;
typedef __attribute__((ext_vector_type(4)))  unsigned int u32x4;
typedef __attribute__((ext_vector_type(8)))  int i32x8;
typedef __attribute__((ext_vector_type(4)))  int i32x4;

__device__ __forceinline__ v16bf cat8(v8bf lo, v8bf hi) {
  return __builtin_shufflevector(lo, hi, 0,1,2,3,4,5,6,7,8,9,10,11,12,13,14,15);
}

// Async copy of one 16-byte chunk global -> LDS (CDNA5 async path, ASYNCcnt).
__device__ __forceinline__ void async_b128(unsigned lds_byte_off, const void* gsrc) {
#if USE_ASYNC
  unsigned long long g = (unsigned long long)gsrc;
  asm volatile("global_load_async_to_lds_b128 %0, %1, off"
               :: "v"(lds_byte_off), "v"(g) : "memory");
#endif
}
__device__ __forceinline__ void async_wait0() {
#if USE_ASYNC
  asm volatile("s_wait_asynccnt 0x0" ::: "memory");
#endif
}
__device__ __forceinline__ unsigned lds_off(const void* p) {
  // low 32 bits of a generic pointer to LDS == LDS byte offset
  return (unsigned)(unsigned long long)p;
}

// ---------------------------------------------------------------------------
// Tensor Data Mover: one-shot 2D tile DMA global -> LDS (bf16 elements).
// D# per CDNA5 ISA:
//   group0: [1:0]=count=1 | [63:32]=lds_addr | [120:64]=global_addr | [127:126]=type=2
//   group1: [17:16]=data_size=1 (2B) | tensor_dim0@[79:48] | tensor_dim1@[111:80]
//           tile_dim0@[127:112] | tile_dim1@[143:128] | tensor_dim0_stride@[207:160]
// Tile rows are packed consecutively into LDS. Issue from ONE wave only.
// ---------------------------------------------------------------------------
__device__ __forceinline__ void tdm_load_2d(unsigned lds_byte, const void* gsrc,
                                            unsigned tile_d0, unsigned tile_d1,
                                            unsigned long long stride0_elems) {
#if HAVE_TDM
  unsigned long long ga = (unsigned long long)gsrc;
  u32x4 g0;
  g0[0] = 1u;                                               // count=1 (user D#)
  g0[1] = lds_byte;                                         // lds_addr
  g0[2] = (unsigned)(ga & 0xffffffffull);                   // global_addr[31:0]
  g0[3] = (unsigned)((ga >> 32) & 0x01ffffffull) | (2u << 30); // addr[56:32] | type=2
  const unsigned td0 = tile_d0, td1 = tile_d1;              // tensor dims = tile dims
  i32x8 g1;
  g1[0] = (int)(1u << 16);                                  // data_size=1 -> 2 bytes
  g1[1] = (int)((td0 & 0xffffu) << 16);                     // tensor_dim0[15:0]
  g1[2] = (int)(((td0 >> 16) & 0xffffu) | ((td1 & 0xffffu) << 16));
  g1[3] = (int)(((td1 >> 16) & 0xffffu) | ((tile_d0 & 0xffffu) << 16));
  g1[4] = (int)(tile_d1 & 0xffffu);                         // tile_dim1 | tile_dim2=0
  g1[5] = (int)(unsigned)(stride0_elems & 0xffffffffull);   // dim0_stride[31:0]
  g1[6] = (int)(unsigned)((stride0_elems >> 32) & 0xffffu); // dim0_stride[47:32]
  g1[7] = 0;
  i32x4 z4 = {0, 0, 0, 0};
#if __clang_major__ >= 23
  i32x8 z8 = {0, 0, 0, 0, 0, 0, 0, 0};
  __builtin_amdgcn_tensor_load_to_lds(g0, g1, z4, z4, z8, 0);
#else
  __builtin_amdgcn_tensor_load_to_lds(g0, g1, z4, z4, 0);
#endif
#else
  (void)lds_byte; (void)gsrc; (void)tile_d0; (void)tile_d1; (void)stride0_elems;
#endif
}
__device__ __forceinline__ void tdm_wait0() {
#if HAVE_TDM
  __builtin_amdgcn_s_wait_tensorcnt(0);
#endif
}

// ---------------------------------------------------------------------------
// f32 -> bf16 elementwise pre-conversion (convert exactly once per element;
// GEMMs then re-stage pure bf16 tiles via async/TDM with no VALU work).
// ---------------------------------------------------------------------------
__global__ __launch_bounds__(256)
void cvt_bf16_kernel(const float* __restrict__ in, bf16_t* __restrict__ out, int n) {
  int i = (blockIdx.x * 256 + threadIdx.x) * 4;
  if (i < n) {
    float4 v = *(const float4*)(in + i);
    out[i + 0] = (bf16_t)v.x;
    out[i + 1] = (bf16_t)v.y;
    out[i + 2] = (bf16_t)v.z;
    out[i + 3] = (bf16_t)v.w;
  }
}

// ---------------------------------------------------------------------------
// Y = X @ W^T + bias, scaled. X bf16 [8192,1024], W bf16 [1024,1024].
// 128x128 block tile, 8 waves, each wave 32x64 (8 WMMAs per K=32 step).
// Double-buffered async LDS staging.
// MODE 0: Y bf16 per-head [B,H,L,DH];  MODE 1: Y bf16 transposed [B,H,DH,L];
// MODE 2: Y f32 plain [8192,1024].
// ---------------------------------------------------------------------------
template <int MODE>
__global__ __launch_bounds__(256)
void gemm_bias_kernel(const bf16_t* __restrict__ X, const bf16_t* __restrict__ W,
                      const float* __restrict__ bias, void* __restrict__ Yv,
                      float scale) {
  __shared__ bf16_t Xs[2][128][32];   // 2 x 8 KB
  __shared__ bf16_t Ws[2][128][32];   // 2 x 8 KB
  const int tid  = threadIdx.x;
  const int lane = tid & 31;
  const int wave = tid >> 5;
  const int row0 = blockIdx.y * 128;
  const int col0 = blockIdx.x * 128;
  const int waveM = wave >> 1;  // 0..3 -> 32 output rows each
  const int waveN = wave & 1;   // 0..1 -> 64 output cols each

  v8f acc[2][4];
  #pragma unroll
  for (int m = 0; m < 2; ++m)
    #pragma unroll
    for (int n = 0; n < 4; ++n) acc[m][n] = (v8f){};

  // stage one K-slab: 128 rows x 32 halfs = 8KB = 512 b128 per buffer
  auto stage = [&](int buf, int kk) {
    const unsigned xb = lds_off(&Xs[buf][0][0]);
    const unsigned wb = lds_off(&Ws[buf][0][0]);
    #pragma unroll
    for (int i = tid; i < 512; i += 256) {
      int r = i >> 2, c = (i & 3) * 8;   // 8 halfs = 16B per chunk
      async_b128(xb + i * 16, X + (size_t)(row0 + r) * D_MODEL + kk + c);
      async_b128(wb + i * 16, W + (size_t)(col0 + r) * D_MODEL + kk + c);
    }
  };

  stage(0, 0);
  async_wait0();
  __syncthreads();

  int buf = 0;
  for (int kk = 0; kk < D_MODEL; kk += 32) {
    if (kk + 32 < D_MODEL) stage(buf ^ 1, kk + 32);  // overlap with compute

    const int kb = (lane >> 4) * 8;
    v16bf af[2];
    #pragma unroll
    for (int m = 0; m < 2; ++m) {
      const int arow = waveM * 32 + m * 16 + (lane & 15);
      af[m] = cat8(*(const v8bf*)&Xs[buf][arow][kb],
                   *(const v8bf*)&Xs[buf][arow][16 + kb]);
    }
    const int kob = (lane >> 4) * 8;
    v16bf bfrag[4];
    #pragma unroll
    for (int n = 0; n < 4; ++n) {
      const int bcol = waveN * 64 + n * 16 + (lane & 15);
      bfrag[n] = cat8(*(const v8bf*)&Ws[buf][bcol][kob],
                      *(const v8bf*)&Ws[buf][bcol][kob + 16]);
    }
    #pragma unroll
    for (int m = 0; m < 2; ++m)
      #pragma unroll
      for (int n = 0; n < 4; ++n)
        acc[m][n] = __builtin_amdgcn_wmma_f32_16x16x32_bf16(
            false, af[m], false, bfrag[n], (short)0, acc[m][n], false, false);

    async_wait0();       // this wave's async stores into buf^1 have landed
    __syncthreads();     // all waves done computing buf / staging buf^1
    buf ^= 1;
  }

  #pragma unroll
  for (int m = 0; m < 2; ++m) {
    const int rbase = row0 + waveM * 32 + m * 16 + ((lane >> 4) << 3);
    #pragma unroll
    for (int n = 0; n < 4; ++n) {
      const int ocol = col0 + waveN * 64 + n * 16 + (lane & 15);
      const float bv = bias[ocol];
      #pragma unroll
      for (int e = 0; e < 8; ++e) {
        int orow = rbase + e;
        float val = (acc[m][n][e] + bv) * scale;
        if (MODE == 2) {
          ((float*)Yv)[(size_t)orow * D_MODEL + ocol] = val;
        } else {
          int b = orow >> 10, li = orow & 1023;
          int h = ocol >> 7,  d  = ocol & 127;
          if (MODE == 0)
            ((bf16_t*)Yv)[(((size_t)b * HEADS + h) * SEQ + li) * DHEAD + d] = (bf16_t)val;
          else
            ((bf16_t*)Yv)[(((size_t)b * HEADS + h) * DHEAD + d) * SEQ + li] = (bf16_t)val;
        }
      }
    }
  }
}

// ---------------------------------------------------------------------------
// Attention: one block per (b*H+h, 32 q rows). Full score rows in LDS.
// K/V tiles staged global->LDS by the Tensor Data Mover (one D# per chunk).
// ---------------------------------------------------------------------------
__global__ __launch_bounds__(256)
void attn_kernel(const bf16_t* __restrict__ qh, const bf16_t* __restrict__ kh,
                 const bf16_t* __restrict__ vT, const unsigned char* __restrict__ mask,
                 bf16_t* __restrict__ ctx, float* __restrict__ wout) {
  __shared__ float  Ssc[32][SEQ];        // 128 KB scores -> probs
  __shared__ bf16_t Qs[32][DHEAD];       // 8 KB
  __shared__ bf16_t Stage[64 * DHEAD];   // 16 KB K/V staging
  __shared__ float  red[32][8];

  const int tid  = threadIdx.x;
  const int lane = tid & 31;
  const int wave = tid >> 5;
  const int bh   = blockIdx.y;           // b*H + h
  const int q0   = blockIdx.x * 32;
  const int b    = bh >> 3;
  const int h    = bh & 7;

  // load Q tile (already scaled by 1/sqrt(DH))
  {
    const uint4* src = (const uint4*)(qh + ((size_t)bh * SEQ + q0) * DHEAD);
    uint4* dst = (uint4*)&Qs[0][0];
    for (int i = tid; i < (32 * DHEAD) / 8; i += 256) dst[i] = src[i];
  }
  __syncthreads();

  const int waveQ = wave >> 2;  // 0..1 -> 16 q rows
  const int waveS = wave & 3;   // phase1: 16 m cols; phase3: 32 d cols

  // ---- Phase 1: S = q . k^T, masked ----
  for (int mc = 0; mc < SEQ; mc += 64) {
    const bf16_t* ksrc = kh + ((size_t)bh * SEQ + mc) * DHEAD;
    if (mc + 64 < SEQ) __builtin_prefetch(ksrc + 64 * DHEAD, 0, 1);
#if HAVE_TDM
    if (wave == 0) {
      // 64 rows x 128 halfs, contiguous rows (stride = DHEAD)
      tdm_load_2d(lds_off(&Stage[0]), ksrc, DHEAD, 64, DHEAD);
      tdm_wait0();
    }
#else
    {
      const unsigned base = lds_off(&Stage[0]);
      for (int i = tid; i < 1024; i += 256)
        async_b128(base + i * 16, (const char*)ksrc + i * 16);
      async_wait0();
    }
#endif
    __syncthreads();

    v8f s = {};
    #pragma unroll
    for (int kk = 0; kk < DHEAD; kk += 32) {
      const int arow = waveQ * 16 + (lane & 15);
      const int kb   = kk + ((lane >> 4) << 3);
      v16bf a = cat8(*(const v8bf*)&Qs[arow][kb],
                     *(const v8bf*)&Qs[arow][kb + 16]);
      const int brow = waveS * 16 + (lane & 15);
      const int kob  = kk + ((lane >> 4) << 3);
      v16bf bb = cat8(*(const v8bf*)&Stage[brow * DHEAD + kob],
                      *(const v8bf*)&Stage[brow * DHEAD + kob + 16]);
      s = __builtin_amdgcn_wmma_f32_16x16x32_bf16(false, a, false, bb, (short)0, s, false, false);
    }
    const int mglob = mc + waveS * 16 + (lane & 15);
    const bool msk  = mask[(size_t)b * SEQ + mglob] != 0;
    #pragma unroll
    for (int e = 0; e < 8; ++e) {
      int r = waveQ * 16 + ((lane >> 4) << 3) + e;
      Ssc[r][mglob] = msk ? -__builtin_inff() : s[e];
    }
    __syncthreads();
  }

  // ---- Phase 2: softmax (8 threads per row, 128 cols each) ----
  {
    const int r = tid >> 3, seg = tid & 7;
    float mx = -3.4e38f;
    for (int j = seg * 128; j < seg * 128 + 128; ++j) mx = fmaxf(mx, Ssc[r][j]);
    red[r][seg] = mx;
    __syncthreads();
    if (seg == 0) {
      float m2 = red[r][0];
      #pragma unroll
      for (int t = 1; t < 8; ++t) m2 = fmaxf(m2, red[r][t]);
      red[r][0] = m2;
    }
    __syncthreads();
    const float rmax = red[r][0];
    float sum = 0.f;
    for (int j = seg * 128; j < seg * 128 + 128; ++j) {
      float p = __expf(Ssc[r][j] - rmax);
      Ssc[r][j] = p;
      sum += p;
    }
    __syncthreads();
    red[r][seg] = sum;
    __syncthreads();
    if (seg == 0) {
      float s2 = 0.f;
      #pragma unroll
      for (int t = 0; t < 8; ++t) s2 += red[r][t];
      red[r][0] = s2;
    }
    __syncthreads();
    const float inv = 1.0f / red[r][0];
    float* wrow = wout + ((size_t)b * SEQ + (q0 + r)) * SEQ;
    const bool isH0 = (h == 0);
    for (int j = seg * 128; j < seg * 128 + 128; ++j) {
      float p = Ssc[r][j] * inv;
      Ssc[r][j] = p;
      if (isH0) wrow[j] = p;   // weights[:,0,:,:] output
    }
    __syncthreads();
  }

  // ---- Phase 3: ctx = P @ V (V staged as [d][m] from vT, strided rows) ----
  v8f o0 = {}; v8f o1 = {};
  for (int mc = 0; mc < SEQ; mc += 64) {
#if HAVE_TDM
    if (wave == 0) {
      // 128 rows (d) x 64 halfs (m), row pitch SEQ elems -> packed [d][64] in LDS
      tdm_load_2d(lds_off(&Stage[0]), vT + (size_t)bh * DHEAD * SEQ + mc,
                  64, DHEAD, SEQ);
      tdm_wait0();
    }
#else
    {
      const unsigned base = lds_off(&Stage[0]);
      for (int i = tid; i < 1024; i += 256) {
        int d = i >> 3, c = i & 7;
        const bf16_t* src = vT + ((size_t)bh * DHEAD + d) * SEQ + mc + c * 8;
        async_b128(base + i * 16, src);
      }
      async_wait0();
    }
#endif
    __syncthreads();

    #pragma unroll
    for (int kk = 0; kk < 64; kk += 32) {
      const int arow = waveQ * 16 + (lane & 15);
      const int cb   = mc + kk + ((lane >> 4) << 3);
      v16bf a;
      #pragma unroll
      for (int j = 0; j < 8; ++j) {
        a[j]     = (bf16_t)Ssc[arow][cb + j];
        a[j + 8] = (bf16_t)Ssc[arow][cb + 16 + j];
      }
      const int d0 = waveS * 32 + (lane & 15);
      const int mo = kk + ((lane >> 4) << 3);
      v16bf b0 = cat8(*(const v8bf*)&Stage[d0 * 64 + mo],
                      *(const v8bf*)&Stage[d0 * 64 + mo + 16]);
      v16bf b1 = cat8(*(const v8bf*)&Stage[(d0 + 16) * 64 + mo],
                      *(const v8bf*)&Stage[(d0 + 16) * 64 + mo + 16]);
      o0 = __builtin_amdgcn_wmma_f32_16x16x32_bf16(false, a, false, b0, (short)0, o0, false, false);
      o1 = __builtin_amdgcn_wmma_f32_16x16x32_bf16(false, a, false, b1, (short)0, o1, false, false);
    }
    __syncthreads();
  }

  #pragma unroll
  for (int e = 0; e < 8; ++e) {
    int r  = q0 + waveQ * 16 + ((lane >> 4) << 3) + e;
    int d0 = waveS * 32 + (lane & 15);
    ctx[((size_t)b * SEQ + r) * D_MODEL + h * DHEAD + d0]      = (bf16_t)o0[e];
    ctx[((size_t)b * SEQ + r) * D_MODEL + h * DHEAD + d0 + 16] = (bf16_t)o1[e];
  }
}

// ---------------------------------------------------------------------------
extern "C" void kernel_launch(void* const* d_in, const int* in_sizes, int n_in,
                              void* d_out, int out_size, void* d_ws, size_t ws_size,
                              hipStream_t stream) {
  const float* query  = (const float*)d_in[0];
  const float* memory = (const float*)d_in[1];
  const unsigned char* mask = (const unsigned char*)d_in[2];  // bool [B,M]
  const float* Wk = (const float*)d_in[3];
  const float* bk = (const float*)d_in[4];
  const float* Wv = (const float*)d_in[5];
  const float* bv = (const float*)d_in[6];
  const float* Wq = (const float*)d_in[7];
  const float* bq = (const float*)d_in[8];
  const float* Wf = (const float*)d_in[9];
  const float* bf_ = (const float*)d_in[10];

  const size_t actElems = (size_t)BATCH * SEQ * D_MODEL;  // 8M
  const size_t wElems   = (size_t)D_MODEL * D_MODEL;      // 1M

  bf16_t* qbf  = (bf16_t*)d_ws;          // query  bf16
  bf16_t* mbf  = qbf + actElems;         // memory bf16
  bf16_t* wkb  = mbf + actElems;         // weights bf16
  bf16_t* wvb  = wkb + wElems;
  bf16_t* wqb  = wvb + wElems;
  bf16_t* wfb  = wqb + wElems;
  bf16_t* qh   = wfb + wElems;           // per-head projections
  bf16_t* kh   = qh + actElems;
  bf16_t* vT   = kh + actElems;
  bf16_t* ctx  = vT + actElems;

  float* outF = (float*)d_out;                        // [B,Q,D_MODEL]
  float* outW = outF + (size_t)BATCH * SEQ * SEQ;     // weights head 0 [B,Q,M]

  const float qscale = 0.08838834764831845f;          // 1/sqrt(128)
  dim3 block(256);

  cvt_bf16_kernel<<<(int)(actElems / 1024), block, 0, stream>>>(query,  qbf, (int)actElems);
  cvt_bf16_kernel<<<(int)(actElems / 1024), block, 0, stream>>>(memory, mbf, (int)actElems);
  cvt_bf16_kernel<<<(int)(wElems / 1024),   block, 0, stream>>>(Wk, wkb, (int)wElems);
  cvt_bf16_kernel<<<(int)(wElems / 1024),   block, 0, stream>>>(Wv, wvb, (int)wElems);
  cvt_bf16_kernel<<<(int)(wElems / 1024),   block, 0, stream>>>(Wq, wqb, (int)wElems);
  cvt_bf16_kernel<<<(int)(wElems / 1024),   block, 0, stream>>>(Wf, wfb, (int)wElems);

  dim3 ggrid(D_MODEL / 128, (BATCH * SEQ) / 128);     // 8 x 64
  gemm_bias_kernel<0><<<ggrid, block, 0, stream>>>(mbf, wkb, bk, kh, 1.0f);
  gemm_bias_kernel<1><<<ggrid, block, 0, stream>>>(mbf, wvb, bv, vT, 1.0f);
  gemm_bias_kernel<0><<<ggrid, block, 0, stream>>>(qbf, wqb, bq, qh, qscale);
  attn_kernel<<<dim3(SEQ / 32, BATCH * HEADS), block, 0, stream>>>(qh, kh, vT, mask, ctx, outW);
  gemm_bias_kernel<2><<<ggrid, block, 0, stream>>>(ctx, wfb, bf_, outF, 1.0f);
}
